// MinimalCausalConceptAttention_60112362275517
// MI455X (gfx1250) — matrix operand order repaired and checked by reading, not compiled, not run on hardware
//
#include <hip/hip_runtime.h>

// ---------------------------------------------------------------------------
// MI455X (gfx1250, wave32) implementation of MinimalCausalConceptAttention.
// All heavy math in bf16 WMMA (v_wmma_f32_16x16x32_bf16) with fp32 accumulate.
// GEMMs: compile-time shapes, double-buffered LDS, global_prefetch_b8 ahead,
// and GLOBAL_LOAD_ASYNC_TO_LDS_B128 for the non-transposed tile copies,
// retired with s_wait_asynccnt.
// ---------------------------------------------------------------------------

typedef __attribute__((ext_vector_type(16))) __bf16 bf16x16;
typedef __attribute__((ext_vector_type(8)))  __bf16 bf16x8;
typedef __attribute__((ext_vector_type(8)))  float  f32x8;
typedef int v4i __attribute__((vector_size(16)));   // builtin's pointee type

#define B_   2
#define T_   2048
#define C_   128
#define D_   1024
#define H_   16
#define HD_  64
#define SCALE_ 0.125f   // 1/sqrt(64)

#if __has_builtin(__builtin_amdgcn_global_load_async_to_lds_b128)
#define HAVE_ASYNC_LDS 1
#endif

#ifdef HAVE_ASYNC_LDS
// async global -> LDS, 16 bytes per lane (tracked by ASYNCcnt, not DScnt)
static __device__ __forceinline__ void async_b128(const __bf16* g, __bf16* l) {
  __builtin_amdgcn_global_load_async_to_lds_b128(
      (__attribute__((address_space(1))) v4i*)(void*)g,
      (__attribute__((address_space(3))) v4i*)(void*)l, 0, 0);
}
static __device__ __forceinline__ void async_wait0() {
#if __has_builtin(__builtin_amdgcn_s_wait_asynccnt)
  __builtin_amdgcn_s_wait_asynccnt(0);
#else
  asm volatile("s_wait_asynccnt 0" ::: "memory");
#endif
}
#endif

static __device__ __forceinline__ __bf16 f2bf(float f) {
  union { float f; unsigned u; } x; x.f = f;
  unsigned r = x.u + 0x7FFFu + ((x.u >> 16) & 1u);   // round-to-nearest-even
  unsigned short h = (unsigned short)(r >> 16);
  union { unsigned short s; __bf16 b; } y; y.s = h;
  return y.b;
}

// Build a v16 bf16 fragment from two 16-byte chunks.
static __device__ __forceinline__ bf16x16 ld_pair(const __bf16* p0, const __bf16* p1) {
  bf16x8 lo = *(const bf16x8*)p0;
  bf16x8 hi = *(const bf16x8*)p1;
  return __builtin_shufflevector(lo, hi, 0,1,2,3,4,5,6,7,8,9,10,11,12,13,14,15);
}
static __device__ __forceinline__ bf16x16 ld32(const __bf16* p) { return ld_pair(p, p + 8); }

static __device__ __forceinline__ f32x8 wmma_bf(bf16x16 a, bf16x16 b, f32x8 c) {
  return __builtin_amdgcn_wmma_f32_16x16x32_bf16(false, a, false, b, (short)0, c, false, false);
}

// ---------------------------------------------------------------------------
// Elementwise helpers (vectorized: 4 floats / thread)
// ---------------------------------------------------------------------------
__global__ void cvt_f32_bf16(const float* __restrict__ in, __bf16* __restrict__ out, long n4) {
  long i = (long)blockIdx.x * blockDim.x + threadIdx.x;
  if (i < n4) {
    float4 v = *(const float4*)(in + i * 4);
    union { __bf16 b[4]; uint2 u; } o;
    o.b[0] = f2bf(v.x); o.b[1] = f2bf(v.y); o.b[2] = f2bf(v.z); o.b[3] = f2bf(v.w);
    *(uint2*)(out + i * 4) = o.u;
  }
}

__global__ void copy_f32(const float* __restrict__ in, float* __restrict__ out, long n4) {
  long i = (long)blockIdx.x * blockDim.x + threadIdx.x;
  if (i < n4) *(float4*)(out + i * 4) = *(const float4*)(in + i * 4);
}

// ---------------------------------------------------------------------------
// bf16 GEMM:  out[M,N] = A[M,K] @ B[K,N] + bias[N]    (N, K compile-time)
// 128x128 block tile, 256 threads (8 waves), wave tile 32x64, K-step 32,
// double-buffered LDS (one barrier per K-step), prefetch 2 steps ahead.
// A-tile copied with async-to-LDS when available; B-tile transposed in regs.
// ---------------------------------------------------------------------------
template<int N, int K, bool F32OUT>
__global__ __launch_bounds__(256)
void gemm_bf16(const __bf16* __restrict__ A, const __bf16* __restrict__ Bm,
               const float* __restrict__ bias, void* __restrict__ outp)
{
  __shared__ __align__(16) __bf16 sA[2][128 * 32];   // A tiles, row-major
  __shared__ __align__(16) __bf16 sBt[2][128 * 32];  // B tiles, transposed [n][k]

  const int tid  = threadIdx.x;
  const int lane = tid & 31, wave = tid >> 5;
  const int half = lane >> 4, ln = lane & 15;
  const int mBase = blockIdx.y * 128, nBase = blockIdx.x * 128;
  const int wm = wave >> 1, wn = wave & 1;

  // per-thread load coordinates (hoisted)
  const int ar = tid >> 1, ac = (tid & 1) * 16;            // A: 128 rows x 32 cols
  const int bk = tid >> 3, bc = (tid & 7) * 16;            // B: 32 rows x 128 cols
  const __bf16* gA = A + (long)(mBase + ar) * K + ac;
  const __bf16* gB = Bm + (long)bk * N + nBase + bc;

  f32x8 acc[2][4];
#pragma unroll
  for (int mi = 0; mi < 2; ++mi)
#pragma unroll
    for (int ni = 0; ni < 4; ++ni) acc[mi][ni] = (f32x8)0.0f;

  auto loadTile = [&](int buf, int k0) {
    const __bf16* g = gA + k0;
    __bf16* la = &sA[buf][ar * 32 + ac];
#ifdef HAVE_ASYNC_LDS
    async_b128(g, la);
    async_b128(g + 8, la + 8);
#else
    *(uint4*)la       = *(const uint4*)g;
    *(uint4*)(la + 8) = *(const uint4*)(g + 8);
#endif
    const __bf16* gb = gB + (long)k0 * N;
    bf16x8 v0 = *(const bf16x8*)gb;
    bf16x8 v1 = *(const bf16x8*)(gb + 8);
#pragma unroll
    for (int i = 0; i < 8; ++i) sBt[buf][(bc + i) * 32 + bk] = v0[i];
#pragma unroll
    for (int i = 0; i < 8; ++i) sBt[buf][(bc + 8 + i) * 32 + bk] = v1[i];
  };

  constexpr int NK = K / 32;
  loadTile(0, 0);

  for (int kb = 0; kb < NK; ++kb) {
    const int buf = kb & 1;
#ifdef HAVE_ASYNC_LDS
    async_wait0();             // retire my async-to-LDS writes from last iter
#endif
    __syncthreads();           // publish tile(kb) to all waves
    if (kb + 1 < NK) loadTile(buf ^ 1, (kb + 1) * 32);
    if (kb + 2 < NK) {         // pull tile kb+2 into GL2 (global_prefetch_b8)
      __builtin_prefetch(gA + (kb + 2) * 32, 0, 1);
      __builtin_prefetch(gB + (long)(kb + 2) * 32 * N, 0, 1);
    }

    bf16x16 af[2], bfr[4];
#pragma unroll
    for (int mi = 0; mi < 2; ++mi) { // A operand: lane=m, K = half*16 + i (contiguous k order)
      int r = wm * 32 + mi * 16 + ln;
      af[mi] = ld32(&sA[buf][r * 32 + half * 16]);
    }
#pragma unroll
    for (int ni = 0; ni < 4; ++ni) { // B operand: lane=n, K = half*16 + i (contiguous k order)
      int n = wn * 64 + ni * 16 + ln;
      bfr[ni] = ld32(&sBt[buf][n * 32 + half * 16]);
    }
#pragma unroll
    for (int mi = 0; mi < 2; ++mi)
#pragma unroll
      for (int ni = 0; ni < 4; ++ni)
        acc[mi][ni] = wmma_bf(af[mi], bfr[ni], acc[mi][ni]);
  }

  // Epilogue (branch-free): C layout: lane holds col n=ln, rows j + 8*half
#pragma unroll
  for (int mi = 0; mi < 2; ++mi)
#pragma unroll
    for (int ni = 0; ni < 4; ++ni) {
      const int gn = nBase + wn * 64 + ni * 16 + ln;
      const float bv = bias[gn];
      const long rbase = (long)(mBase + wm * 32 + mi * 16 + 8 * half) * N + gn;
#pragma unroll
      for (int j = 0; j < 8; ++j) {
        float v = acc[mi][ni][j] + bv;
        if (F32OUT) ((float*)outp)[rbase + (long)j * N] = v;
        else        ((__bf16*)outp)[rbase + (long)j * N] = f2bf(v);
      }
    }
}

// ---------------------------------------------------------------------------
// Flash-attention phase. MASKMODE 0: causal (k <= q); 1: concept (k <= q>>4).
// One wave owns 16 query rows; key blocks of 32 staged in LDS by the block.
// K tile copied async-to-LDS when available; V transposed through registers.
// ---------------------------------------------------------------------------
template<int MASKMODE>
static __device__ void flash_phase(const __bf16* __restrict__ Kbase,
                                   const __bf16* __restrict__ Vbase,
                                   int rowStride, int nkb, int q0, int tid,
                                   const bf16x16* qf,
                                   __bf16* sK, __bf16* sVt, __bf16* sP,
                                   f32x8* Ofr, float* mrow, float* lrow)
{
  const int lane = tid & 31, wave = tid >> 5;
  const int half = lane >> 4, ln = lane & 15;
  __bf16* sPw = sP + wave * (16 * 32);

#pragma unroll
  for (int t = 0; t < 4; ++t) Ofr[t] = (f32x8)0.0f;
#pragma unroll
  for (int j = 0; j < 8; ++j) { mrow[j] = -1e30f; lrow[j] = 0.0f; }

  for (int kb = 0; kb < nkb; ++kb) {
    const int kBase = kb * 32;
    __syncthreads();           // everyone done reading previous K/V tiles
    { // K tile 32x64 row-major; V tile transposed [vdim][key]
      int kk = tid >> 2, d0 = (tid & 3) * 16;
      const __bf16* g = Kbase + (long)(kBase + kk) * rowStride + d0;
      __bf16* lk = &sK[kk * 64 + d0];
#ifdef HAVE_ASYNC_LDS
      async_b128(g, lk);
      async_b128(g + 8, lk + 8);
#else
      *(uint4*)lk       = *(const uint4*)g;
      *(uint4*)(lk + 8) = *(const uint4*)(g + 8);
#endif
      const __bf16* gv = Vbase + (long)(kBase + kk) * rowStride + d0;
      bf16x8 v0 = *(const bf16x8*)gv;
      bf16x8 v1 = *(const bf16x8*)(gv + 8);
#pragma unroll
      for (int i = 0; i < 8; ++i) sVt[(d0 + i) * 32 + kk] = v0[i];
#pragma unroll
      for (int i = 0; i < 8; ++i) sVt[(d0 + 8 + i) * 32 + kk] = v1[i];
    }
#ifdef HAVE_ASYNC_LDS
    async_wait0();             // K-tile async writes landed in LDS
#endif
    __syncthreads();           // publish K/V tiles

    // S = Q (16x64) x K^T : two key sub-tiles of 16, K-chunks of 32
    f32x8 s0 = (f32x8)0.0f, s1 = (f32x8)0.0f;
#pragma unroll
    for (int c = 0; c < 2; ++c) {
      bf16x16 b0 = ld32(&sK[ln * 64        + c * 32 + half * 16]);
      bf16x16 b1 = ld32(&sK[(16 + ln) * 64 + c * 32 + half * 16]);
      s0 = wmma_bf(qf[c], b0, s0);
      s1 = wmma_bf(qf[c], b1, s1);
    }

    const int k0g = kBase + ln, k1g = kBase + 16 + ln;
    float pv0[8], pv1[8];
#pragma unroll
    for (int j = 0; j < 8; ++j) {
      int qg  = q0 + j + 8 * half;
      int lim = (MASKMODE == 0) ? qg : (qg >> 4);
      bool ok0 = (k0g <= lim), ok1 = (k1g <= lim);
      float a0 = ok0 ? s0[j] * SCALE_ : -1e30f;
      float a1 = ok1 ? s1[j] * SCALE_ : -1e30f;
      float mx = fmaxf(a0, a1);
#pragma unroll
      for (int off = 1; off < 16; off <<= 1) mx = fmaxf(mx, __shfl_xor(mx, off, 32));
      float mn = fmaxf(mrow[j], mx);
      float p0 = ok0 ? __expf(a0 - mn) : 0.0f;   // explicit 0 for masked lanes
      float p1 = ok1 ? __expf(a1 - mn) : 0.0f;
      pv0[j] = p0; pv1[j] = p1;
      float ps = p0 + p1;
#pragma unroll
      for (int off = 1; off < 16; off <<= 1) ps += __shfl_xor(ps, off, 32);
      float alpha = __expf(mrow[j] - mn);
      lrow[j] = lrow[j] * alpha + ps;
      mrow[j] = mn;
#pragma unroll
      for (int t = 0; t < 4; ++t) Ofr[t][j] *= alpha;
    }

    // C-layout -> A-layout redistribution of P through LDS (per-wave region)
#pragma unroll
    for (int j = 0; j < 8; ++j) {
      sPw[(j + 8 * half) * 32 + ln]      = f2bf(pv0[j]);
      sPw[(j + 8 * half) * 32 + 16 + ln] = f2bf(pv1[j]);
    }
    __syncthreads();

    bf16x16 pf = ld32(&sPw[ln * 32 + half * 16]);
#pragma unroll
    for (int t = 0; t < 4; ++t) {
      bf16x16 vf = ld32(&sVt[(t * 16 + ln) * 32 + half * 16]);
      Ofr[t] = wmma_bf(pf, vf, Ofr[t]);
    }
  }
}

// grid: (T/64, H, B), 128 threads (4 waves x 16 query rows)
__global__ __launch_bounds__(128)
void attn_kernel(const __bf16* __restrict__ all_t,
                 const __bf16* __restrict__ all_c,
                 __bf16* __restrict__ xout)
{
  __shared__ __align__(16) __bf16 sK[32 * 64];
  __shared__ __align__(16) __bf16 sVt[64 * 32];
  __shared__ __align__(16) __bf16 sP[4 * 16 * 32];

  const int tid = threadIdx.x;
  const int lane = tid & 31, wave = tid >> 5;
  const int half = lane >> 4, ln = lane & 15;
  const int qBase = blockIdx.x * 64;
  const int h = blockIdx.y, b = blockIdx.z;
  const int q0 = qBase + wave * 16;

  // Q fragments (kept in registers for both phases)
  const __bf16* Qb = all_t + (long)b * T_ * 3072 + h * HD_;
  bf16x16 qf[2];
  {
    const __bf16* qr = Qb + (long)(q0 + ln) * 3072;
#pragma unroll
    for (int c = 0; c < 2; ++c) {
      const __bf16* p = qr + c * 32 + half * 16;
      qf[c] = ld32(p);
    }
  }

  f32x8 Ot[4], Oc[4];
  float mt[8], lt[8], mc[8], lc[8];

  // Phase 1: causal token-token (keys up to qBase+63 -> 2*(bx+1) key blocks)
  const __bf16* Kt = all_t + (long)b * T_ * 3072 + 1024 + h * HD_;
  const __bf16* Vt = all_t + (long)b * T_ * 3072 + 2048 + h * HD_;
  flash_phase<0>(Kt, Vt, 3072, 2 * ((int)blockIdx.x + 1), q0, tid, qf, sK, sVt, sP, Ot, mt, lt);

  // Phase 2: token-to-concept (all C=128 concepts, masked)
  const __bf16* Kc = all_c + (long)b * C_ * 2048 + h * HD_;
  const __bf16* Vc = all_c + (long)b * C_ * 2048 + 1024 + h * HD_;
  flash_phase<1>(Kc, Vc, 2048, C_ / 32, q0, tid, qf, sK, sVt, sP, Oc, mc, lc);

  // x = xtt + xtc  (separate softmax normalizations, then sum)
#pragma unroll
  for (int t = 0; t < 4; ++t) {
    int vcol = t * 16 + ln;
#pragma unroll
    for (int j = 0; j < 8; ++j) {
      int qg = q0 + j + 8 * half;
      float v = Ot[t][j] / lt[j] + Oc[t][j] / lc[j];
      xout[(long)(b * T_ + qg) * D_ + h * HD_ + vcol] = f2bf(v);
    }
  }
}

// ---------------------------------------------------------------------------
// Host-side orchestration
// ---------------------------------------------------------------------------
extern "C" void kernel_launch(void* const* d_in, const int* in_sizes, int n_in,
                              void* d_out, int out_size, void* d_ws, size_t ws_size,
                              hipStream_t stream)
{
  const float* xt = (const float*)d_in[0];
  const float* xc = (const float*)d_in[1];
  const float* Wt = (const float*)d_in[2];
  const float* bt = (const float*)d_in[3];
  const float* Wc = (const float*)d_in[4];
  const float* bc = (const float*)d_in[5];
  const float* Wp = (const float*)d_in[6];
  const float* bp = (const float*)d_in[7];
  float* out = (float*)d_out;

  // Workspace carve-up (bf16 elements), ~54 MB total
  __bf16* w = (__bf16*)d_ws;
  __bf16* xt_bf = w;  w += (long)4096 * 1024;
  __bf16* Wt_bf = w;  w += (long)1024 * 3072;
  __bf16* allt  = w;  w += (long)4096 * 3072;
  __bf16* xc_bf = w;  w += (long)256 * 1024;
  __bf16* Wc_bf = w;  w += (long)1024 * 2048;
  __bf16* allc  = w;  w += (long)256 * 2048;
  __bf16* xat   = w;  w += (long)4096 * 1024;
  __bf16* Wp_bf = w;

  auto cvt = [&](const float* src, __bf16* dst, long n) {
    long n4 = n / 4;
    cvt_f32_bf16<<<(int)((n4 + 255) / 256), 256, 0, stream>>>(src, dst, n4);
  };
  cvt(xt, xt_bf, (long)4096 * 1024);
  cvt(Wt, Wt_bf, (long)1024 * 3072);
  cvt(xc, xc_bf, (long)256 * 1024);
  cvt(Wc, Wc_bf, (long)1024 * 2048);
  cvt(Wp, Wp_bf, (long)1024 * 1024);

  // all_t = xt @ Wt + bt   [4096 x 3072]   (Q|K|V packed per row)
  gemm_bf16<3072, 1024, false><<<dim3(3072 / 128, 4096 / 128), 256, 0, stream>>>(
      xt_bf, Wt_bf, bt, allt);
  // all_c = xc @ Wc + bc   [256 x 2048]    (Kc|Vc packed per row)
  gemm_bf16<2048, 1024, false><<<dim3(2048 / 128, 256 / 128), 256, 0, stream>>>(
      xc_bf, Wc_bf, bc, allc);

  // dual flash attention -> xat [B*T, D] bf16
  attn_kernel<<<dim3(T_ / 64, H_, B_), 128, 0, stream>>>(allt, allc, xat);

  // xt_out = xat @ Wp + bp -> fp32 directly into d_out
  gemm_bf16<1024, 1024, true><<<dim3(1024 / 128, 4096 / 128), 256, 0, stream>>>(
      xat, Wp_bf, bp, out);

  // second tuple output: xc passthrough (vectorized)
  copy_f32<<<(int)((((long)256 * 1024) / 4 + 255) / 256), 256, 0, stream>>>(
      xc, out + (long)4096 * 1024, (long)256 * 1024 / 4);
}
